// MambaUpLayer_14224931685037
// MI455X (gfx1250) — compile-verified, matrix operand order checked by reading, not run on hardware
//
#include <hip/hip_runtime.h>
#include <hip/hip_bf16.h>

// ---------------- CDNA5 WMMA fragment helpers (wave32) ----------------
typedef _Float16 v16h __attribute__((ext_vector_type(16)));
typedef _Float16 v8h  __attribute__((ext_vector_type(8)));
typedef float    v8f  __attribute__((ext_vector_type(8)));

#define DIMC   96
#define DIN    192
#define NSTATE 16
#define SEQL   16384   // 128*128
#define NB     2

__device__ __forceinline__ int lane_id() { return threadIdx.x & 31; }

// A fragment: row-major [M x K] fp32 weights, tile origin (m0, kb).
// ISA 16-bit A 16x32 layout: lane L(0-15) row=L holds K {kb..kb+7, kb+16..kb+23};
// lane L+16 holds K {kb+8..kb+15, kb+24..kb+31}.
__device__ __forceinline__ v16h loadA(const float* __restrict__ A, int M, int K,
                                      int m0, int kb) {
  const int lane = lane_id();
  const int row  = m0 + (lane & 15);
  const int kh   = kb + ((lane >> 4) << 3);
  v16h f;
  if (row < M) {
    const float* p = A + (size_t)row * K + kh;
#pragma unroll
    for (int i = 0; i < 8; ++i) {
      f[i]     = (_Float16)p[i];
      f[8 + i] = (_Float16)p[16 + i];
    }
  } else {
#pragma unroll
    for (int i = 0; i < 16; ++i) f[i] = (_Float16)0.f;
  }
  return f;
}

// B fragment from f16 transposed activations Bt[L][K]: lane holds column (=l row of Bt),
// elements contiguous in K -> two 16-byte vector loads per lane.
__device__ __forceinline__ v16h loadBt(const _Float16* __restrict__ Bt, int K,
                                       int row, int kh) {
  const _Float16* p = Bt + (size_t)row * K + kh;
  const v8h lo = *(const v8h*)(p);
  const v8h hi = *(const v8h*)(p + 16);
  v16h f;
#pragma unroll
  for (int i = 0; i < 8; ++i) { f[i] = lo[i]; f[8 + i] = hi[i]; }
  return f;
}

// C/D store: VGPR r -> row m0 + r + (lane>>4)*8, col n0 + (lane&15)
__device__ __forceinline__ void storeD(float* __restrict__ Dm, int M, int L,
                                       int m0, int n0, const v8f& c,
                                       const float* __restrict__ bias,
                                       const float* __restrict__ res) {
  const int lane = lane_id();
  const int col  = n0 + (lane & 15);
  const int rb   = m0 + ((lane >> 4) << 3);
#pragma unroll
  for (int r = 0; r < 8; ++r) {
    const int row = rb + r;
    if (row < M) {
      float v = c[r];
      if (bias) v += bias[row];
      if (res)  v += res[(size_t)row * L + col];
      Dm[(size_t)row * L + col] = v;
    }
  }
}

#define WMMA32(A_, B_, C_) \
  __builtin_amdgcn_wmma_f32_16x16x32_f16(false, (A_), false, (B_), (short)0, (C_), false, false)

// ---- Generic GEMM v2: A fp32 [M,K] x Bt f16 [L,K] -> D fp32 [M,L], 16x64 per wave ----
__global__ void gemm_cm2_kernel(const float* __restrict__ A, int M, int K,
                                const _Float16* __restrict__ Bt, long long strideBt,
                                float* __restrict__ Dm, long long strideD,
                                const float* __restrict__ bias,
                                const float* __restrict__ res, long long strideR,
                                int L, int nBatch) {
  const int mtiles = (M + 15) >> 4;
  const int ntiles = L >> 6;                      // 64-wide N strips
  const int tpb    = mtiles * ntiles;
  const int tile   = blockIdx.x * (blockDim.x >> 5) + (threadIdx.x >> 5);
  if (tile >= tpb * nBatch) return;
  const int b  = tile / tpb;
  const int t  = tile % tpb;
  const int m0 = (t / ntiles) << 4;
  const int n0 = (t % ntiles) << 6;
  const _Float16* Bb = Bt + (long long)b * strideBt;
  const int lane = lane_id();
  const int colx = lane & 15;
  v8f a0 = {}, a1 = {}, a2 = {}, a3 = {};
  for (int kb = 0; kb < K; kb += 32) {
    const v16h af = loadA(A, M, K, m0, kb);
    const int kh  = kb + ((lane >> 4) << 3);
    const v16h b0 = loadBt(Bb, K, n0 +  0 + colx, kh);
    const v16h b1 = loadBt(Bb, K, n0 + 16 + colx, kh);
    const v16h b2 = loadBt(Bb, K, n0 + 32 + colx, kh);
    const v16h b3 = loadBt(Bb, K, n0 + 48 + colx, kh);
    a0 = WMMA32(af, b0, a0);
    a1 = WMMA32(af, b1, a1);
    a2 = WMMA32(af, b2, a2);
    a3 = WMMA32(af, b3, a3);
  }
  float* Db = Dm + (long long)b * strideD;
  const float* rb = res ? res + (long long)b * strideR : nullptr;
  storeD(Db, M, L, m0, n0 +  0, a0, bias, rb);
  storeD(Db, M, L, m0, n0 + 16, a1, bias, rb);
  storeD(Db, M, L, m0, n0 + 32, a2, bias, rb);
  storeD(Db, M, L, m0, n0 + 48, a3, bias, rb);
}

// convert fp32 channel-major [C][L] -> f16 transposed [L][C] (per batch)
__global__ void cvt_tr_kernel(const float* __restrict__ src, long long strideS,
                              _Float16* __restrict__ dst, long long strideDst,
                              int C, int L, int nBatch) {
  const long long idx = (long long)blockIdx.x * blockDim.x + threadIdx.x;
  const long long per = (long long)C * L;
  if (idx >= per * nBatch) return;
  const int b = (int)(idx / per);
  const long long r = idx - (long long)b * per;
  const int l = (int)(r / C), c = (int)(r - (long long)l * C);
  dst[(long long)b * strideDst + (long long)l * C + c] =
      (_Float16)src[(long long)b * strideS + (size_t)c * L + l];
}

// ---------------- conv1: 1x1 over concat(upsampled x, Hx), K = 288 ----------------
__device__ __forceinline__ v16h loadB_cat(const float* __restrict__ up,
                                          const float* __restrict__ Hx,
                                          int n0, int kb) {
  const int lane = lane_id();
  const int col  = n0 + (lane & 15);
  const int kh   = kb + ((lane >> 4) << 3);
  v16h f;
#pragma unroll
  for (int i = 0; i < 8; ++i) {
    const int k1 = kh + i, k2 = kh + 16 + i;
    f[i]     = (_Float16)((k1 < DIMC) ? up[(size_t)k1 * SEQL + col]
                                      : Hx[(size_t)(k1 - DIMC) * SEQL + col]);
    f[8 + i] = (_Float16)((k2 < DIMC) ? up[(size_t)k2 * SEQL + col]
                                      : Hx[(size_t)(k2 - DIMC) * SEQL + col]);
  }
  return f;
}

__global__ void gemm_conv1_kernel(const float* __restrict__ W,
                                  const float* __restrict__ bias,
                                  const float* __restrict__ up,
                                  const float* __restrict__ Hx,
                                  float* __restrict__ out) {
  const int ntiles = SEQL >> 4, mtiles = DIMC >> 4;      // 1024, 6
  const int tpb = mtiles * ntiles;
  const int tile = blockIdx.x * (blockDim.x >> 5) + (threadIdx.x >> 5);
  if (tile >= tpb * NB) return;
  const int b = tile / tpb, t = tile % tpb;
  const int m0 = (t / ntiles) << 4, n0 = (t % ntiles) << 4;
  const float* upb = up + (size_t)b * DIMC * SEQL;
  const float* hxb = Hx + (size_t)b * 2 * DIMC * SEQL;
  v8f acc = {};
  for (int kb = 0; kb < 3 * DIMC; kb += 32) {            // K = 288
    v16h af = loadA(W, DIMC, 3 * DIMC, m0, kb);
    v16h bf = loadB_cat(upb, hxb, n0, kb);
    acc = WMMA32(af, bf, acc);
  }
  storeD(out + (size_t)b * DIMC * SEQL, DIMC, SEQL, m0, n0, acc, bias, nullptr);
}

// ---------------- conv2: 3x3 implicit GEMM, K = 96*9 = 864 ----------------
__device__ __forceinline__ _Float16 conv2_fetch(const float* __restrict__ h1,
                                                int kk, int col) {
  const int ci  = kk / 9;
  const int tap = kk - ci * 9;
  const int h   = (col >> 7) + (tap / 3) - 1;
  const int w   = (col & 127) + (tap % 3) - 1;
  if ((unsigned)h < 128u && (unsigned)w < 128u)
    return (_Float16)h1[(size_t)ci * SEQL + (h << 7) + w];
  return (_Float16)0.f;
}

__global__ void gemm_conv2_kernel(const float* __restrict__ W,
                                  const float* __restrict__ bias,
                                  const float* __restrict__ h1,
                                  float* __restrict__ out) {
  const int ntiles = SEQL >> 4, mtiles = DIMC >> 4;
  const int tpb = mtiles * ntiles;
  const int tile = blockIdx.x * (blockDim.x >> 5) + (threadIdx.x >> 5);
  if (tile >= tpb * NB) return;
  const int b = tile / tpb, t = tile % tpb;
  const int m0 = (t / ntiles) << 4, n0 = (t % ntiles) << 4;
  const float* h1b = h1 + (size_t)b * DIMC * SEQL;
  const int lane = lane_id();
  const int col  = n0 + (lane & 15);
  v8f acc = {};
  for (int kb = 0; kb < DIMC * 9; kb += 32) {            // 27 WMMA steps
    v16h af = loadA(W, DIMC, DIMC * 9, m0, kb);
    const int kh = kb + ((lane >> 4) << 3);
    v16h bf;
#pragma unroll
    for (int i = 0; i < 8; ++i) {
      bf[i]     = conv2_fetch(h1b, kh + i, col);
      bf[8 + i] = conv2_fetch(h1b, kh + 16 + i, col);
    }
    acc = WMMA32(af, bf, acc);
  }
  storeD(out + (size_t)b * DIMC * SEQL, DIMC, SEQL, m0, n0, acc, bias, nullptr);
}

// ---------------- elementwise / reduction kernels ----------------
__global__ void upsample_kernel(const float* __restrict__ x, float* __restrict__ up) {
  const int idx = blockIdx.x * blockDim.x + threadIdx.x;
  if (idx >= NB * DIMC * SEQL) return;
  const int l = idx & (SEQL - 1), bc = idx >> 14;
  const int oh = l >> 7, ow = l & 127;
  const float fr = oh * (63.f / 127.f), fc = ow * (63.f / 127.f);
  const int r0 = (int)fr, c0 = (int)fc;
  const int r1 = min(r0 + 1, 63), c1 = min(c0 + 1, 63);
  const float rw = fr - r0, cw = fc - c0;
  const float* p = x + (size_t)bc * 4096;
  const float v00 = p[r0 * 64 + c0], v01 = p[r0 * 64 + c1];
  const float v10 = p[r1 * 64 + c0], v11 = p[r1 * 64 + c1];
  up[idx] = (v00 * (1.f - cw) + v01 * cw) * (1.f - rw)
          + (v10 * (1.f - cw) + v11 * cw) * rw;
}

__global__ void stats_kernel(const float* __restrict__ h, float* __restrict__ st) {
  const int bc = blockIdx.x;                              // 0..191
  const float* p = h + (size_t)bc * SEQL;
  float s = 0.f, s2 = 0.f;
  for (int i = threadIdx.x; i < SEQL; i += blockDim.x) {
    const float v = p[i]; s += v; s2 += v * v;
  }
  __shared__ float ss[256], sq[256];
  ss[threadIdx.x] = s; sq[threadIdx.x] = s2;
  __syncthreads();
  for (int o = 128; o > 0; o >>= 1) {
    if (threadIdx.x < o) { ss[threadIdx.x] += ss[threadIdx.x + o];
                           sq[threadIdx.x] += sq[threadIdx.x + o]; }
    __syncthreads();
  }
  if (threadIdx.x == 0) {
    const float m = ss[0] * (1.f / SEQL);
    const float v = sq[0] * (1.f / SEQL) - m * m;
    st[bc * 2] = m; st[bc * 2 + 1] = rsqrtf(v + 1e-5f);
  }
}

__global__ void norm_selu_kernel(float* __restrict__ h, const float* __restrict__ st) {
  const int idx = blockIdx.x * blockDim.x + threadIdx.x;
  if (idx >= NB * DIMC * SEQL) return;
  const int bc = idx >> 14;
  const float v = (h[idx] - st[bc * 2]) * st[bc * 2 + 1];
  const float sc = 1.0507009873554805f, al = 1.6732632423543772f;
  h[idx] = v > 0.f ? sc * v : sc * al * (__expf(v) - 1.f);
}

__global__ void conv1d_silu_kernel(const float* __restrict__ xz,
                                   const float* __restrict__ w,
                                   const float* __restrict__ bias,
                                   float* __restrict__ u) {
  const int idx = blockIdx.x * blockDim.x + threadIdx.x;
  if (idx >= NB * DIN * SEQL) return;
  const int l = idx & (SEQL - 1), bd = idx >> 14;
  const int b = bd / DIN, d = bd - b * DIN;
  const float* xin = xz + ((size_t)b * 2 * DIN + d) * SEQL;   // rows 0..191 = xin
  float s = bias[d];
#pragma unroll
  for (int t = 0; t < 4; ++t) {
    const int ll = l + t - 3;
    if (ll >= 0) s += w[d * 4 + t] * xin[ll];
  }
  u[idx] = s / (1.f + __expf(-s));                            // SiLU
}

__global__ void dtproj_kernel(const float* __restrict__ xdbl,
                              const float* __restrict__ Wd,
                              const float* __restrict__ bd,
                              float* __restrict__ dtf) {
  const int idx = blockIdx.x * blockDim.x + threadIdx.x;
  if (idx >= NB * DIN * SEQL) return;
  const int l = idx & (SEQL - 1), bdp = idx >> 14;
  const int b = bdp / DIN, d = bdp - b * DIN;
  const float* xb = xdbl + (size_t)b * 48 * SEQL;
  float s = bd[d];
#pragma unroll
  for (int r = 0; r < 6; ++r) s += Wd[d * 6 + r] * xb[(size_t)r * SEQL + l];
  dtf[idx] = (s > 20.f) ? s : log1pf(__expf(s));              // softplus
}

// Selective scan: one 16-lane group per (b,d); lane = state index n.
__global__ void scan_kernel(const float* __restrict__ u,
                            const float* __restrict__ dtf,
                            const float* __restrict__ xdbl,
                            const float* __restrict__ A_log,
                            const float* __restrict__ Dv,
                            float* __restrict__ ys) {
  const int gid = (blockIdx.x * blockDim.x + threadIdx.x) >> 4;
  const int n   = threadIdx.x & 15;
  if (gid >= NB * DIN) return;
  const int b = gid / DIN, d = gid - b * DIN;
  const float A  = -__expf(A_log[d * NSTATE + n]);
  const float Dd = Dv[d];
  const float* up  = u   + ((size_t)b * DIN + d) * SEQL;
  const float* dtp = dtf + ((size_t)b * DIN + d) * SEQL;
  const float* Bp  = xdbl + ((size_t)b * 48 + 6  + n) * SEQL;
  const float* Cp  = xdbl + ((size_t)b * 48 + 22 + n) * SEQL;
  float* yp = ys + ((size_t)b * DIN + d) * SEQL;
  float h = 0.f;
  for (int l = 0; l < SEQL; ++l) {
    if ((l & 63) == 0) {                                     // global_prefetch_b8
      __builtin_prefetch(Bp + l + 256, 0, 1);
      __builtin_prefetch(Cp + l + 256, 0, 1);
      __builtin_prefetch(up + l + 256, 0, 1);
      __builtin_prefetch(dtp + l + 256, 0, 1);
    }
    const float dt = dtp[l], uu = up[l];
    h = h * __expf(dt * A) + (dt * uu) * Bp[l];
    float y = h * Cp[l];
    y += __shfl_xor(y, 1, 32);                               // reduce 16 states
    y += __shfl_xor(y, 2, 32);
    y += __shfl_xor(y, 4, 32);
    y += __shfl_xor(y, 8, 32);
    if (n == 0) yp[l] = y + uu * Dd;
  }
}

__global__ void gate_kernel(float* __restrict__ y, const float* __restrict__ xz) {
  const int idx = blockIdx.x * blockDim.x + threadIdx.x;
  if (idx >= NB * DIN * SEQL) return;
  const int l = idx & (SEQL - 1), bd = idx >> 14;
  const int b = bd / DIN, d = bd - b * DIN;
  const float z = xz[((size_t)b * 2 * DIN + DIN + d) * SEQL + l];
  y[idx] *= z / (1.f + __expf(-z));
}

// ---------------- driver ----------------
extern "C" void kernel_launch(void* const* d_in, const int* in_sizes, int n_in,
                              void* d_out, int out_size, void* d_ws, size_t ws_size,
                              hipStream_t stream) {
  const float* x         = (const float*)d_in[0];
  const float* Hx        = (const float*)d_in[1];
  const float* conv1_w   = (const float*)d_in[2];
  const float* conv1_b   = (const float*)d_in[3];
  const float* conv2_w   = (const float*)d_in[4];
  const float* conv2_b   = (const float*)d_in[5];
  const float* conv3_w   = (const float*)d_in[6];
  const float* conv3_b   = (const float*)d_in[7];
  const float* in_proj_w = (const float*)d_in[8];
  const float* conv1d_w  = (const float*)d_in[9];
  const float* conv1d_b  = (const float*)d_in[10];
  const float* x_proj_w  = (const float*)d_in[11];
  const float* dt_proj_w = (const float*)d_in[12];
  const float* dt_proj_b = (const float*)d_in[13];
  const float* A_log     = (const float*)d_in[14];
  const float* Dv        = (const float*)d_in[15];
  const float* out_proj_w= (const float*)d_in[16];
  float* out = (float*)d_out;

  const size_t CL  = (size_t)DIMC * SEQL;       // 96*16384 per batch
  const size_t nCL = (size_t)NB * CL;
  float* ws   = (float*)d_ws;
  float* bufA = ws;                 // up -> later h2
  float* bufB = bufA + nCL;         // h1 -> later h3 / Lx (fp32)
  float* xz   = bufB + nCL;         // [B][384][L]
  float* ubuf = xz   + (size_t)NB * 2 * DIN * SEQL;   // [B][192][L]
  float* xdbl = ubuf + (size_t)NB * DIN * SEQL;       // [B][48][L]
  float* dtf  = xdbl + (size_t)NB * 48 * SEQL;        // [B][192][L]
  float* ysc  = dtf  + (size_t)NB * DIN * SEQL;       // [B][192][L]
  float* st   = ysc  + (size_t)NB * DIN * SEQL;       // [192][2] (+pad)
  // f16 transposed staging buffers [L][K] per batch
  _Float16* h2t = (_Float16*)(st + 512);
  _Float16* lxt = h2t + (size_t)NB * SEQL * DIMC;
  _Float16* ut  = lxt + (size_t)NB * SEQL * DIMC;
  _Float16* yt  = ut  + (size_t)NB * SEQL * DIN;

  const int EW = 256;
  const int nElemC = (int)nCL;                        // 2*96*16384
  const int nElemD = NB * DIN * SEQL;                 // 2*192*16384
  const long long sC   = (long long)CL;
  const long long sXZ  = (long long)2 * DIN * SEQL;
  const long long sU   = (long long)DIN * SEQL;
  const long long sXD  = (long long)48 * SEQL;
  const long long sTc  = (long long)SEQL * DIMC;      // f16 [L][96] batch stride
  const long long sTd  = (long long)SEQL * DIN;       // f16 [L][192] batch stride

  const int nt16 = SEQL / 16;                         // 1024 (conv1/conv2)
  const int nt64 = SEQL / 64;                         // 256  (gemm v2)
  auto blocks = [](int tiles) { return (tiles + 3) / 4; };

  // 1) upsample x -> bufA
  upsample_kernel<<<(nElemC + EW - 1) / EW, EW, 0, stream>>>(x, bufA);
  // 2) conv1 (WMMA, concat gather) -> bufB
  gemm_conv1_kernel<<<blocks(NB * 6 * nt16), 128, 0, stream>>>(conv1_w, conv1_b, bufA, Hx, bufB);
  // 3) conv2 3x3 (WMMA implicit GEMM) -> bufA
  gemm_conv2_kernel<<<blocks(NB * 6 * nt16), 128, 0, stream>>>(conv2_w, conv2_b, bufB, bufA);
  // 3b) h2 -> f16 [L][96]
  cvt_tr_kernel<<<(nElemC + EW - 1) / EW, EW, 0, stream>>>(bufA, sC, h2t, sTc, DIMC, SEQL, NB);
  // 4) conv3 1x1 (WMMA v2, 4 N-tiles/wave) -> bufB
  gemm_cm2_kernel<<<blocks(NB * 6 * nt64), 128, 0, stream>>>(
      conv3_w, DIMC, DIMC, h2t, sTc, bufB, sC, conv3_b, nullptr, 0, SEQL, NB);
  // 5) instance-norm stats (LDS reduction)
  stats_kernel<<<NB * DIMC, 256, 0, stream>>>(bufB, st);
  // 6) normalize + SELU in place -> Lx in bufB
  norm_selu_kernel<<<(nElemC + EW - 1) / EW, EW, 0, stream>>>(bufB, st);
  // 6b) Lx -> f16 [L][96]
  cvt_tr_kernel<<<(nElemC + EW - 1) / EW, EW, 0, stream>>>(bufB, sC, lxt, sTc, DIMC, SEQL, NB);
  // 7) in_proj (WMMA v2): xz[384,L] = in_proj_w[384,96] x Lx[96,L]
  gemm_cm2_kernel<<<blocks(NB * 24 * nt64), 128, 0, stream>>>(
      in_proj_w, 2 * DIN, DIMC, lxt, sTc, xz, sXZ, nullptr, nullptr, 0, SEQL, NB);
  // 8) causal depthwise conv1d + SiLU -> u
  conv1d_silu_kernel<<<(nElemD + EW - 1) / EW, EW, 0, stream>>>(xz, conv1d_w, conv1d_b, ubuf);
  // 8b) u -> f16 [L][192]
  cvt_tr_kernel<<<(nElemD + EW - 1) / EW, EW, 0, stream>>>(ubuf, sU, ut, sTd, DIN, SEQL, NB);
  // 9) x_proj (WMMA v2, M=38 padded to 48): xdbl = x_proj_w[38,192] x u[192,L]
  gemm_cm2_kernel<<<blocks(NB * 3 * nt64), 128, 0, stream>>>(
      x_proj_w, 38, DIN, ut, sTd, xdbl, sXD, nullptr, nullptr, 0, SEQL, NB);
  // 10) dt_proj + softplus -> dtf
  dtproj_kernel<<<(nElemD + EW - 1) / EW, EW, 0, stream>>>(xdbl, dt_proj_w, dt_proj_b, dtf);
  // 11) selective scan (16 lanes = 16 states per (b,d))
  scan_kernel<<<(NB * DIN * 16 + 255) / 256, 256, 0, stream>>>(ubuf, dtf, xdbl, A_log, Dv, ysc);
  // 12) gate: y *= silu(z)
  gate_kernel<<<(nElemD + EW - 1) / EW, EW, 0, stream>>>(ysc, xz);
  // 12b) gated y -> f16 [L][192]
  cvt_tr_kernel<<<(nElemD + EW - 1) / EW, EW, 0, stream>>>(ysc, sU, yt, sTd, DIN, SEQL, NB);
  // 13) out_proj (WMMA v2) + residual Lx -> d_out
  gemm_cm2_kernel<<<blocks(NB * 6 * nt64), 128, 0, stream>>>(
      out_proj_w, DIMC, DIN, yt, sTd, out, sC, nullptr, bufB, sC, SEQL, NB);
}
